// SelfAttnTest_14980845929064
// MI455X (gfx1250) — matrix operand order repaired and checked by reading, not compiled, not run on hardware
//
#include <hip/hip_runtime.h>

// ---------------------------------------------------------------------------
// Conformer rel-pos multi-head self-attention for gfx1250 (MI455X).
// bf16 WMMA (v_wmma_f32_16x16x32_bf16) for every GEMM stage; flash-style
// attention with the rel_shift folded into per-tile diagonal indexing
// (scores_bd[q,k] = q_v[q] . p[k-q+T-1]) so the [T,2T-1] tensor is never
// materialized.  V-slab staging uses gfx1250 async global->LDS copies
// (global_load_async_to_lds_b128 / s_wait_asynccnt).
// ---------------------------------------------------------------------------

#define BSZ   8
#define TSEQ  1024
#define DMODEL 512
#define NHEAD 8
#define DK    64
#define MROWS (BSZ * TSEQ)        // 8192
#define NPOS  (2 * TSEQ - 1)      // 2047
#define NPOS_PAD 2048
#define SCALE 0.125f              // 1/sqrt(64)

typedef __attribute__((ext_vector_type(16))) __bf16 v16bf;
typedef __attribute__((ext_vector_type(8)))  __bf16 v8bf;
typedef __attribute__((ext_vector_type(8)))  float  v8f;

static __device__ __forceinline__ v8f vzero8() {
  v8f z = {0.f, 0.f, 0.f, 0.f, 0.f, 0.f, 0.f, 0.f};
  return z;
}

static __device__ __forceinline__ v8f wmma_bf16(v16bf a, v16bf b, v8f c) {
  // D = A*B + C, fp32 accumulate
  return __builtin_amdgcn_wmma_f32_16x16x32_bf16(
      false, a, false, b, (short)0, c, false, false);
}

// A-fragment (16x32 bf16, MxK): lane L holds row M = L%16; with half = L>>4,
// element e covers K = half*16+e.
// Loaded as one contiguous 32B chunk per lane from a row-major row.
static __device__ __forceinline__ v16bf load_a16(const __bf16* row, int kk, int half) {
  return *(const v16bf*)(row + kk + half * 16);
}

// half-wave (16-lane) row reductions; wave32 only.
static __device__ __forceinline__ float rmax16(float v) {
  v = fmaxf(v, __shfl_xor(v, 1, 16));
  v = fmaxf(v, __shfl_xor(v, 2, 16));
  v = fmaxf(v, __shfl_xor(v, 4, 16));
  v = fmaxf(v, __shfl_xor(v, 8, 16));
  return v;
}
static __device__ __forceinline__ float rsum16(float v) {
  v += __shfl_xor(v, 1, 16);
  v += __shfl_xor(v, 2, 16);
  v += __shfl_xor(v, 4, 16);
  v += __shfl_xor(v, 8, 16);
  return v;
}

// gfx1250 async global -> LDS copy (16B per lane), tracked by ASYNCcnt.
static __device__ __forceinline__ void async_copy_b128(unsigned lds_addr,
                                                       const __bf16* gaddr) {
  asm volatile("global_load_async_to_lds_b128 %0, %1, off"
               :
               : "v"(lds_addr), "v"((unsigned long long)(uintptr_t)gaddr)
               : "memory");
}
static __device__ __forceinline__ void wait_asynccnt0() {
  asm volatile("s_wait_asynccnt 0x0" ::: "memory");
}

// ---------------------------------------------------------------------------
// Kernel 0: LayerNorm -> bf16  (one block of 256 per token row)
// ---------------------------------------------------------------------------
__global__ __launch_bounds__(256) void ln_kernel(const float* __restrict__ x,
                                                 const float* __restrict__ gamma,
                                                 const float* __restrict__ beta,
                                                 __bf16* __restrict__ y) {
  const int row = blockIdx.x;
  const int tid = threadIdx.x;
  const float* xr = x + (size_t)row * DMODEL;
  float a = xr[tid];
  float b = xr[tid + 256];
  float s = a + b;
  float ss = a * a + b * b;
#pragma unroll
  for (int m = 1; m < 32; m <<= 1) {
    s += __shfl_xor(s, m, 32);
    ss += __shfl_xor(ss, m, 32);
  }
  __shared__ float S[8], SS[8];
  const int w = tid >> 5;
  if ((tid & 31) == 0) { S[w] = s; SS[w] = ss; }
  __syncthreads();
  float st = 0.f, sst = 0.f;
#pragma unroll
  for (int i = 0; i < 8; ++i) { st += S[i]; sst += SS[i]; }
  const float mu = st * (1.f / DMODEL);
  const float var = sst * (1.f / DMODEL) - mu * mu;
  const float rinv = rsqrtf(var + 1e-5f);
  __bf16* yr = y + (size_t)row * DMODEL;
  yr[tid]       = (__bf16)((a - mu) * rinv * gamma[tid] + beta[tid]);
  yr[tid + 256] = (__bf16)((b - mu) * rinv * gamma[tid + 256] + beta[tid + 256]);
}

// ---------------------------------------------------------------------------
// Kernel 1: weight transpose + f32->bf16   Wt[n][k] = W[k][n], 512x512
// ---------------------------------------------------------------------------
__global__ __launch_bounds__(256) void wtrans_kernel(const float* __restrict__ W,
                                                     __bf16* __restrict__ Wt) {
  const int idx = blockIdx.x * 256 + threadIdx.x;   // < 512*512
  const int k = idx >> 9;
  const int n = idx & 511;
  Wt[(size_t)n * DMODEL + k] = (__bf16)W[idx];
}

// f32 -> bf16 copy with zero padding past n_src
__global__ __launch_bounds__(256) void cvt_kernel(const float* __restrict__ src,
                                                  __bf16* __restrict__ dst,
                                                  int n_src) {
  const int idx = blockIdx.x * 256 + threadIdx.x;
  dst[idx] = (idx < n_src) ? (__bf16)src[idx] : (__bf16)0.f;
}

// ---------------------------------------------------------------------------
// Kernel 2: fused QKV projection.  One wave computes a 16(M) x 64(N) tile for
// all three weights.  All 12 B-fragments of a K-step are loaded into distinct
// registers first (one load clause / one wait), then the 12 WMMAs issue
// back-to-back so next-iteration loads overlap matrix-core work.
// grid = (M/16, 2), block = 128 (4 waves, each owns one 64-wide N block)
// ---------------------------------------------------------------------------
__global__ __launch_bounds__(128) void proj_qkv_kernel(
    const __bf16* __restrict__ y,
    const __bf16* __restrict__ wqt, const __bf16* __restrict__ wkt,
    const __bf16* __restrict__ wvt,
    const float* __restrict__ biasq, const float* __restrict__ biask,
    const float* __restrict__ biasv,
    const float* __restrict__ pbu, const float* __restrict__ pbv,
    __bf16* __restrict__ qu, __bf16* __restrict__ qv,
    __bf16* __restrict__ kmat, __bf16* __restrict__ vmat) {
  const int wave = threadIdx.x >> 5;
  const int L = threadIdx.x & 31;
  const int half = L >> 4, nl = L & 15;
  const int m0 = blockIdx.x * 16;
  const int n0 = (blockIdx.y * 4 + wave) * 64;

  const __bf16* arow = y + (size_t)(m0 + nl) * DMODEL;
  const __bf16* wqrow[4];
  const __bf16* wkrow[4];
  const __bf16* wvrow[4];
#pragma unroll
  for (int ct = 0; ct < 4; ++ct) {
    const size_t wo = (size_t)(n0 + ct * 16 + nl) * DMODEL + half * 16;
    wqrow[ct] = wqt + wo;
    wkrow[ct] = wkt + wo;
    wvrow[ct] = wvt + wo;
  }

  v8f accq[4], acck[4], accv[4];
#pragma unroll
  for (int i = 0; i < 4; ++i) { accq[i] = vzero8(); acck[i] = vzero8(); accv[i] = vzero8(); }

  for (int kk = 0; kk < DMODEL; kk += 32) {
    if (kk + 32 < DMODEL) __builtin_prefetch(arow + kk + 32, 0, 0);
    v16bf af = load_a16(arow, kk, half);
    v16bf fq[4], fk[4], fv[4];
#pragma unroll
    for (int ct = 0; ct < 4; ++ct) {
      fq[ct] = *(const v16bf*)(wqrow[ct] + kk);
      fk[ct] = *(const v16bf*)(wkrow[ct] + kk);
      fv[ct] = *(const v16bf*)(wvrow[ct] + kk);
    }
#pragma unroll
    for (int ct = 0; ct < 4; ++ct) {
      accq[ct] = wmma_bf16(af, fq[ct], accq[ct]);
      acck[ct] = wmma_bf16(af, fk[ct], acck[ct]);
      accv[ct] = wmma_bf16(af, fv[ct], accv[ct]);
    }
  }

#pragma unroll
  for (int ct = 0; ct < 4; ++ct) {
    const int n = n0 + ct * 16 + nl;
    const int h = n >> 6, dk = n & 63;
    const float bqv = biasq[n], bkv = biask[n], bvv = biasv[n];
    const float uv = pbu[n], vv = pbv[n];
#pragma unroll
    for (int r = 0; r < 8; ++r) {
      const int m = m0 + half * 8 + r;
      const int bb = m >> 10, t = m & 1023;
      const size_t o = (((size_t)(bb * NHEAD + h)) * TSEQ + t) * DK + dk;
      const float qval = accq[ct][r] + bqv;
      qu[o] = (__bf16)(qval + uv);
      qv[o] = (__bf16)(qval + vv);
      kmat[o] = (__bf16)(acck[ct][r] + bkv);
      vmat[o] = (__bf16)(accv[ct][r] + bvv);
    }
  }
}

// ---------------------------------------------------------------------------
// Kernel 3: positional projection p = pos_emb @ Wp (no bias), row-major bf16.
// grid = (NPOS_PAD/16, 2), block = 128
// ---------------------------------------------------------------------------
__global__ __launch_bounds__(128) void proj_p_kernel(const __bf16* __restrict__ pe,
                                                     const __bf16* __restrict__ wpt,
                                                     __bf16* __restrict__ pmat) {
  const int wave = threadIdx.x >> 5;
  const int L = threadIdx.x & 31;
  const int half = L >> 4, nl = L & 15;
  const int m0 = blockIdx.x * 16;
  const int n0 = (blockIdx.y * 4 + wave) * 64;

  const __bf16* arow = pe + (size_t)(m0 + nl) * DMODEL;
  v8f acc[4];
#pragma unroll
  for (int i = 0; i < 4; ++i) acc[i] = vzero8();

  for (int kk = 0; kk < DMODEL; kk += 32) {
    v16bf af = load_a16(arow, kk, half);
    v16bf fb[4];
#pragma unroll
    for (int ct = 0; ct < 4; ++ct)
      fb[ct] = *(const v16bf*)(wpt + (size_t)(n0 + ct * 16 + nl) * DMODEL + kk + half * 16);
#pragma unroll
    for (int ct = 0; ct < 4; ++ct)
      acc[ct] = wmma_bf16(af, fb[ct], acc[ct]);
  }
#pragma unroll
  for (int ct = 0; ct < 4; ++ct) {
    const int n = n0 + ct * 16 + nl;
#pragma unroll
    for (int r = 0; r < 8; ++r) {
      const int m = m0 + half * 8 + r;
      if (m < NPOS) pmat[(size_t)m * DMODEL + n] = (__bf16)acc[ct][r];
    }
  }
}

// ---------------------------------------------------------------------------
// Kernel 4: flash-style rel-pos attention. One wave per (b,h, 16-row q tile),
// streaming 32 keys per step.  BD term uses the folded rel_shift:
//   bd[q0+dq, k0+dc] = G[dq][15 + dc - dq],  G = q_v_tile x p[base-15 .. base+16]
// with base = k0 - q0 + T - 1 (diagonal gather through LDS).  V slab staged
// with async global->LDS B128 copies (ASYNCcnt), overlapped with score work.
// ---------------------------------------------------------------------------
#define VROW 72   // bf16 elems per padded V row (144B: 16B aligned, conflict-free)

__global__ __launch_bounds__(32) void attn_kernel(
    const __bf16* __restrict__ qu, const __bf16* __restrict__ qv,
    const __bf16* __restrict__ kmat, const __bf16* __restrict__ vmat,
    const __bf16* __restrict__ pmat, __bf16* __restrict__ attn_out) {
  const int bh = blockIdx.x;            // b*8 + h
  const int h = bh & 7;
  const int bb = bh >> 3;
  const int q0 = blockIdx.y * 16;
  const int L = threadIdx.x & 31;
  const int half = L >> 4, nl = L & 15;

  __shared__ float  Gs[16][33];         // 16x32 BD pre-shift tile (+pad)
  __shared__ __bf16 Ps[16][32];         // probabilities tile (A operand)
  __shared__ __bf16 Vs[32][VROW];       // staged V slab (async dest)

  const unsigned vs_lds = (unsigned)(uintptr_t)(&Vs[0][0]) + (unsigned)(L * VROW * 2);

  const __bf16* qurow = qu + ((size_t)bh * TSEQ + q0 + nl) * DK;
  const __bf16* qvrow = qv + ((size_t)bh * TSEQ + q0 + nl) * DK;
  const v16bf au0 = load_a16(qurow, 0, half);
  const v16bf au1 = load_a16(qurow, 32, half);
  const v16bf av0 = load_a16(qvrow, 0, half);
  const v16bf av1 = load_a16(qvrow, 32, half);

  const __bf16* kbase = kmat + (size_t)bh * TSEQ * DK;
  const __bf16* vbase = vmat + (size_t)bh * TSEQ * DK;

  float Mrow[8], Lrow[8];
  v8f O[4];
#pragma unroll
  for (int r = 0; r < 8; ++r) { Mrow[r] = -1e30f; Lrow[r] = 0.f; }
#pragma unroll
  for (int i = 0; i < 4; ++i) O[i] = vzero8();

  for (int kstep = 0; kstep < TSEQ; kstep += 32) {
    // ---- async-stage V slab (lane L copies key row kstep+L, 8 x 16B) ----
    {
      const __bf16* src = vbase + (size_t)(kstep + L) * DK;
#pragma unroll
      for (int i = 0; i < 8; ++i)
        async_copy_b128(vs_lds + (unsigned)(i * 16), src + i * 8);
    }

    float sv[2][8];
#pragma unroll
    for (int sub = 0; sub < 2; ++sub) {
      const int k0 = kstep + sub * 16;
      // ---- AC = (q+u) . k^T ----
      v8f ac = vzero8();
      {
        const __bf16* kcol = kbase + (size_t)(k0 + nl) * DK;
        v16bf kb0 = *(const v16bf*)(kcol + half * 16);
        v16bf kb1 = *(const v16bf*)(kcol + 32 + half * 16);
        ac = wmma_bf16(au0, kb0, ac);
        ac = wmma_bf16(au1, kb1, ac);
      }
      // ---- BD: G = (q+v) . p_slab^T over 32 diagonal-window columns ----
      const int base = k0 - q0 + (TSEQ - 1);
#pragma unroll
      for (int jt = 0; jt < 2; ++jt) {
        v8f g = vzero8();
        int n = base - 15 + jt * 16 + nl;
        if (n > NPOS - 1) n = NPOS - 1;       // j=31 column unused
        const __bf16* prow = pmat + (size_t)n * DMODEL + h * DK;
        v16bf p0 = *(const v16bf*)(prow + half * 16);
        v16bf p1 = *(const v16bf*)(prow + 32 + half * 16);
        g = wmma_bf16(av0, p0, g);
        g = wmma_bf16(av1, p1, g);
#pragma unroll
        for (int r = 0; r < 8; ++r)
          Gs[half * 8 + r][jt * 16 + nl] = g[r];
      }
      __syncthreads();
#pragma unroll
      for (int r = 0; r < 8; ++r) {
        const int row = half * 8 + r;
        sv[sub][r] = (ac[r] + Gs[row][15 + nl - row]) * SCALE;
      }
      __syncthreads();   // Gs reused by next sub-tile
    }

    // ---- online softmax over this 32-key block ----
    float osc[8];
#pragma unroll
    for (int r = 0; r < 8; ++r) {
      float mt = rmax16(fmaxf(sv[0][r], sv[1][r]));
      const float mn = fmaxf(Mrow[r], mt);
      const float sc = __expf(Mrow[r] - mn);
      const float p0 = __expf(sv[0][r] - mn);
      const float p1 = __expf(sv[1][r] - mn);
      Lrow[r] = Lrow[r] * sc + rsum16(p0 + p1);
      Mrow[r] = mn;
      osc[r] = sc;
      Ps[half * 8 + r][nl] = (__bf16)p0;
      Ps[half * 8 + r][16 + nl] = (__bf16)p1;
    }
#pragma unroll
    for (int ct = 0; ct < 4; ++ct)
#pragma unroll
      for (int r = 0; r < 8; ++r) O[ct][r] *= osc[r];
    wait_asynccnt0();   // V slab resident in LDS
    __syncthreads();

    // ---- O += P @ V  (A frag from Ps, B frag gathered from Vs) ----
    const v16bf pa = load_a16(&Ps[nl][0], 0, half);
#pragma unroll
    for (int ct = 0; ct < 4; ++ct) {
      v16bf vf;
#pragma unroll
      for (int e = 0; e < 16; ++e)
        vf[e] = Vs[half * 16 + e][ct * 16 + nl];
      O[ct] = wmma_bf16(pa, vf, O[ct]);
    }
    __syncthreads();   // Vs/Ps reads drained before next-iter overwrite
  }

  // ---- normalize and store bf16 in [B, T, H*DK] for the output GEMM ----
#pragma unroll
  for (int ct = 0; ct < 4; ++ct) {
#pragma unroll
    for (int r = 0; r < 8; ++r) {
      const int t = q0 + half * 8 + r;
      const int col = h * DK + ct * 16 + nl;
      attn_out[((size_t)bb * TSEQ + t) * DMODEL + col] =
          (__bf16)(O[ct][r] * (1.f / Lrow[r]));
    }
  }
}

// ---------------------------------------------------------------------------
// Kernel 5: output projection  out = attn @ Wo + bo  (fp32 result)
// grid = (M/16, 2), block = 128
// ---------------------------------------------------------------------------
__global__ __launch_bounds__(128) void proj_out_kernel(const __bf16* __restrict__ a,
                                                       const __bf16* __restrict__ wot,
                                                       const float* __restrict__ bo,
                                                       float* __restrict__ out) {
  const int wave = threadIdx.x >> 5;
  const int L = threadIdx.x & 31;
  const int half = L >> 4, nl = L & 15;
  const int m0 = blockIdx.x * 16;
  const int n0 = (blockIdx.y * 4 + wave) * 64;

  const __bf16* arow = a + (size_t)(m0 + nl) * DMODEL;
  v8f acc[4];
#pragma unroll
  for (int i = 0; i < 4; ++i) acc[i] = vzero8();

  for (int kk = 0; kk < DMODEL; kk += 32) {
    if (kk + 32 < DMODEL) __builtin_prefetch(arow + kk + 32, 0, 0);
    v16bf af = load_a16(arow, kk, half);
    v16bf fb[4];
#pragma unroll
    for (int ct = 0; ct < 4; ++ct)
      fb[ct] = *(const v16bf*)(wot + (size_t)(n0 + ct * 16 + nl) * DMODEL + kk + half * 16);
#pragma unroll
    for (int ct = 0; ct < 4; ++ct)
      acc[ct] = wmma_bf16(af, fb[ct], acc[ct]);
  }
#pragma unroll
  for (int ct = 0; ct < 4; ++ct) {
    const int n = n0 + ct * 16 + nl;
    const float bias = bo[n];
#pragma unroll
    for (int r = 0; r < 8; ++r) {
      const int m = m0 + half * 8 + r;
      out[(size_t)m * DMODEL + n] = acc[ct][r] + bias;
    }
  }
}

// ---------------------------------------------------------------------------
extern "C" void kernel_launch(void* const* d_in, const int* in_sizes, int n_in,
                              void* d_out, int out_size, void* d_ws, size_t ws_size,
                              hipStream_t stream) {
  (void)in_sizes; (void)n_in; (void)out_size; (void)ws_size;
  const float* x     = (const float*)d_in[0];
  const float* gamma = (const float*)d_in[1];
  const float* beta  = (const float*)d_in[2];
  const float* Wq    = (const float*)d_in[3];
  const float* bq    = (const float*)d_in[4];
  const float* Wk    = (const float*)d_in[5];
  const float* bk    = (const float*)d_in[6];
  const float* Wv    = (const float*)d_in[7];
  const float* bv    = (const float*)d_in[8];
  const float* Wp    = (const float*)d_in[9];
  const float* Wo    = (const float*)d_in[10];
  const float* bo    = (const float*)d_in[11];
  const float* pbu   = (const float*)d_in[12];
  const float* pbv   = (const float*)d_in[13];
  const float* pos   = (const float*)d_in[14];
  float* out = (float*)d_out;

  // workspace carve-up (bf16 buffers, 256B aligned)
  size_t off = 0;
  auto take = [&](size_t elems) -> __bf16* {
    void* p = (char*)d_ws + off;
    off += (elems * sizeof(__bf16) + 255) & ~(size_t)255;
    return (__bf16*)p;
  };
  __bf16* y_bf  = take((size_t)MROWS * DMODEL);
  __bf16* wqt   = take((size_t)DMODEL * DMODEL);
  __bf16* wkt   = take((size_t)DMODEL * DMODEL);
  __bf16* wvt   = take((size_t)DMODEL * DMODEL);
  __bf16* wot   = take((size_t)DMODEL * DMODEL);
  __bf16* wpt   = take((size_t)DMODEL * DMODEL);
  __bf16* pe_bf = take((size_t)NPOS_PAD * DMODEL);
  __bf16* p_bf  = take((size_t)NPOS_PAD * DMODEL);
  __bf16* qu    = take((size_t)MROWS * DMODEL);
  __bf16* qv    = take((size_t)MROWS * DMODEL);
  __bf16* kmat  = take((size_t)MROWS * DMODEL);
  __bf16* vmat  = take((size_t)MROWS * DMODEL);
  __bf16* a_bf  = take((size_t)MROWS * DMODEL);

  // 0) layernorm -> bf16 activations
  ln_kernel<<<MROWS, 256, 0, stream>>>(x, gamma, beta, y_bf);

  // 1) weight transpose+convert, pos_emb convert (padded with zeros)
  const int wblocks = (DMODEL * DMODEL) / 256;
  wtrans_kernel<<<wblocks, 256, 0, stream>>>(Wq, wqt);
  wtrans_kernel<<<wblocks, 256, 0, stream>>>(Wk, wkt);
  wtrans_kernel<<<wblocks, 256, 0, stream>>>(Wv, wvt);
  wtrans_kernel<<<wblocks, 256, 0, stream>>>(Wp, wpt);
  wtrans_kernel<<<wblocks, 256, 0, stream>>>(Wo, wot);
  cvt_kernel<<<(NPOS_PAD * DMODEL) / 256, 256, 0, stream>>>(pos, pe_bf, NPOS * DMODEL);

  // 2) fused QKV projection (+pos biases) into [B,H,T,DK] bf16
  proj_qkv_kernel<<<dim3(MROWS / 16, 2), 128, 0, stream>>>(
      y_bf, wqt, wkt, wvt, bq, bk, bv, pbu, pbv, qu, qv, kmat, vmat);

  // 3) positional projection
  proj_p_kernel<<<dim3(NPOS_PAD / 16, 2), 128, 0, stream>>>(pe_bf, wpt, p_bf);

  // 4) flash rel-pos attention (async V staging)
  attn_kernel<<<dim3(BSZ * NHEAD, TSEQ / 16), 32, 0, stream>>>(
      qu, qv, kmat, vmat, p_bf, a_bf);

  // 5) output projection (fp32 out)
  proj_out_kernel<<<dim3(MROWS / 16, 2), 128, 0, stream>>>(a_bf, wot, bo, out);
}
